// Net_88235808129626
// MI455X (gfx1250) — matrix operand order repaired and checked by reading, not compiled, not run on hardware
//
#include <hip/hip_runtime.h>

typedef __attribute__((ext_vector_type(16))) _Float16 v16h;
typedef __attribute__((ext_vector_type(8)))  _Float16 v8h;
typedef __attribute__((ext_vector_type(4)))  _Float16 v4h;
typedef __attribute__((ext_vector_type(8)))  float    v8f;

#define K_IN    32000
#define N_HID   1000
#define N_OUT   5
#define BATCH   256
#define STEPS   200
#define BETA    0.95f
#define THRESH  1.0f
#define HPAD    1024   // hidden dim padded to multiple of 32 for WMMA K-loop

// ---------------------------------------------------------------------------
// Zero persistent state (mem1 [256x1000], mem2 [256x5]) each call.
// ---------------------------------------------------------------------------
__global__ void zero_state(float* __restrict__ mem1, float* __restrict__ mem2) {
    int i = blockIdx.x * blockDim.x + threadIdx.x;
    if (i < BATCH * N_HID) mem1[i] = 0.f;
    if (i < BATCH * N_OUT) mem2[i] = 0.f;
}

// ---------------------------------------------------------------------------
// One-pass f32 -> f16 conversion of x (32.8 MB read, 16.4 MB write).
// Removes all A-operand conversion VALU from the GEMM inner loop.
// ---------------------------------------------------------------------------
__global__ __launch_bounds__(256)
void convert_x(const float* __restrict__ x, _Float16* __restrict__ xh) {
    const int i = (blockIdx.x * blockDim.x + threadIdx.x) * 4;
    if (i < BATCH * K_IN) {
        float4 v = *(const float4*)(x + i);
        v4h h;
        h[0] = (_Float16)v.x; h[1] = (_Float16)v.y;
        h[2] = (_Float16)v.z; h[3] = (_Float16)v.w;
        *(v4h*)(xh + i) = h;
    }
}

// ---------------------------------------------------------------------------
// GEMM1: cur1[b,h] = sum_k x[b,k] * W1[h,k] + b1[h]
// Grid: 63 blocks (one per 16-wide N tile) x 512 threads (16 waves).
// Wave w owns the 16x16 output tile at M = w*16, so W1 is streamed from HBM
// exactly once (128 MB); f16 x tiles come from L2. A fragments are two
// contiguous b128 f16 loads; only W1 is converted in-loop (once per element).
// ---------------------------------------------------------------------------
__global__ __launch_bounds__(512)
void gemm1_wmma(const _Float16* __restrict__ xh, const float* __restrict__ W1,
                const float* __restrict__ b1, float* __restrict__ cur1) {
    const int tid  = threadIdx.x;
    const int wave = tid >> 5;               // 0..15 -> batch tile
    const int lane = tid & 31;
    const int n0   = blockIdx.x * 16;        // 0..62 -> n0 up to 992
    const int m0   = wave * 16;

    const int mrow = lane & 15;              // A: M row / B,D: N col
    const int kh   = lane >> 4;              // lane half selects K group

    const _Float16* arow = xh + (size_t)(m0 + mrow) * K_IN;
    int nglob = n0 + mrow;
    if (nglob > N_HID - 1) nglob = N_HID - 1;          // branchless clamp
    const float* brow = W1 + (size_t)nglob * K_IN;

    v8f c = {};
    for (int k0 = 0; k0 < K_IN; k0 += 32) {
        // A 16x32 f16: lane half kh holds K = kh*16 .. kh*16+15 (matches B)
        v8h alo = *(const v8h*)(arow + k0 + kh * 16);
        v8h ahi = *(const v8h*)(arow + k0 + kh * 16 + 8);
        v16h a = __builtin_shufflevector(alo, ahi,
                 0,1,2,3,4,5,6,7,8,9,10,11,12,13,14,15);
        // B 32x16: lane half kh holds K = kh*16 .. kh*16+15 (f32, convert here)
        v16h bf;
#pragma unroll
        for (int j = 0; j < 8; ++j) {
            float2 vb = *(const float2*)(brow + k0 + kh * 16 + 2 * j);
            bf[2*j] = (_Float16)vb.x;  bf[2*j+1] = (_Float16)vb.y;
        }
        if (wave == 0)                               // stream W1 ~2KB ahead
            __builtin_prefetch(brow + k0 + 512, 0, 3);
        c = __builtin_amdgcn_wmma_f32_16x16x32_f16(false, a, false, bf,
                                                   (short)0, c, false, false);
    }

    // D layout: VGPR r, lanes 0-15 -> M=r, lanes 16-31 -> M=8+r; N=lane&15
    const int nc = lane & 15;
    if (n0 + nc < N_HID) {
        const float bias = b1[n0 + nc];
#pragma unroll
        for (int r = 0; r < 8; ++r) {
            const int m = kh * 8 + r;
            cur1[(m0 + m) * N_HID + n0 + nc] = c[r] + bias;
        }
    }
}

// ---------------------------------------------------------------------------
// One SNN step, hidden layer: update mem1, emit spikes into LDS as f16,
// then wave 0 computes cur2 tile = spk1[16x1024] @ W2^T[1024x16] via WMMA.
// Grid: 16 blocks (one per 16-batch tile) x 256 threads.
// ---------------------------------------------------------------------------
__global__ __launch_bounds__(256)
void step_hidden(const float* __restrict__ cur1, float* __restrict__ mem1,
                 const float* __restrict__ W2, float* __restrict__ cur2) {
    __shared__ __align__(32) _Float16 spk[16][HPAD];     // 32 KB
    __shared__ __align__(32) _Float16 w2s[N_OUT][HPAD];  // 10 KB, zero padded K
    const int tid = threadIdx.x;
    const int m0  = blockIdx.x * 16;

    // stage W2 (f32 -> f16), zero-pad K 1000..1023
    for (int idx = tid; idx < N_OUT * HPAD; idx += 256) {
        const int o = idx >> 10, k = idx & (HPAD - 1);
        w2s[o][k] = (k < N_HID) ? (_Float16)W2[o * N_HID + k] : (_Float16)0.f;
    }
    // zero-pad spike K tail
    for (int idx = tid; idx < 16 * (HPAD - N_HID); idx += 256) {
        const int b = idx / (HPAD - N_HID);
        const int k = N_HID + idx % (HPAD - N_HID);
        spk[b][k] = (_Float16)0.f;
    }
    // Leaky integrate-and-fire update (subtract reset), spike -> LDS f16
    for (int idx = tid; idx < 16 * N_HID; idx += 256) {
        const int b = idx / N_HID, h = idx % N_HID;
        const int g = (m0 + b) * N_HID + h;
        float m = mem1[g];
        const float reset = (m > THRESH) ? THRESH : 0.f;
        m = BETA * m + cur1[g] - reset;
        mem1[g] = m;
        spk[b][h] = (_Float16)((m > THRESH) ? 1.f : 0.f);
    }
    __syncthreads();

    if (tid < 32) {                 // wave 0 only, EXEC all-ones inside
        const int lane = tid;
        const int mrow = lane & 15;
        const int kh   = lane >> 4;
        const int ncol = lane & 15;
        const bool nvalid = (ncol < N_OUT);
        const _Float16* brow = &w2s[nvalid ? ncol : 0][0];
        const v16h zero16 = {};
        v8f c = {};
        for (int k0 = 0; k0 < HPAD; k0 += 32) {
            v8h alo = *(const v8h*)&spk[mrow][k0 + kh * 16];
            v8h ahi = *(const v8h*)&spk[mrow][k0 + kh * 16 + 8];
            v16h a = __builtin_shufflevector(alo, ahi,
                     0,1,2,3,4,5,6,7,8,9,10,11,12,13,14,15);
            v16h bf = *(const v16h*)&brow[k0 + kh * 16];
            bf = nvalid ? bf : zero16;               // zero columns n>=5
            c = __builtin_amdgcn_wmma_f32_16x16x32_f16(false, a, false, bf,
                                                       (short)0, c, false, false);
        }
#pragma unroll
        for (int r = 0; r < 8; ++r) {
            const int m = kh * 8 + r;
            cur2[(m0 + m) * 16 + ncol] = c[r];   // padded [256][16] scratch
        }
    }
}

// ---------------------------------------------------------------------------
// One SNN step, output layer: mem2 update + write spk2_rec / mem2_rec.
// ---------------------------------------------------------------------------
__global__ void step_out(const float* __restrict__ cur2, const float* __restrict__ b2,
                         float* __restrict__ mem2, float* __restrict__ out, int t) {
    const int i = blockIdx.x * blockDim.x + threadIdx.x;
    if (i >= BATCH * N_OUT) return;
    const int b = i / N_OUT, o = i % N_OUT;
    const float c = cur2[b * 16 + o] + b2[o];
    float m = mem2[i];
    const float reset = (m > THRESH) ? THRESH : 0.f;
    m = BETA * m + c - reset;
    mem2[i] = m;
    out[t * (BATCH * N_OUT) + i] = (m > THRESH) ? 1.f : 0.f;             // spk2_rec
    out[STEPS * BATCH * N_OUT + t * (BATCH * N_OUT) + i] = m;            // mem2_rec
}

// ---------------------------------------------------------------------------
extern "C" void kernel_launch(void* const* d_in, const int* in_sizes, int n_in,
                              void* d_out, int out_size, void* d_ws, size_t ws_size,
                              hipStream_t stream) {
    const float* x  = (const float*)d_in[0];   // [256,2,80,200]
    const float* W1 = (const float*)d_in[1];   // [1000,32000]
    const float* b1 = (const float*)d_in[2];   // [1000]
    const float* W2 = (const float*)d_in[3];   // [5,1000]
    const float* b2 = (const float*)d_in[4];   // [5]
    float* out = (float*)d_out;                // [200,256,5] + [200,256,5]

    float* ws   = (float*)d_ws;
    float* cur1 = ws;                          // 256000 f32
    float* mem1 = ws + BATCH * N_HID;          // 256000 f32
    float* cur2 = ws + 2 * BATCH * N_HID;      // 256*16 f32 (padded)
    float* mem2 = cur2 + BATCH * 16;           // 256*5 f32
    _Float16* xh = (_Float16*)(mem2 + BATCH * N_OUT);  // 256*32000 f16 = 16.4 MB
    (void)in_sizes; (void)n_in; (void)out_size; (void)ws_size;

    zero_state<<<(BATCH * N_HID + 255) / 256, 256, 0, stream>>>(mem1, mem2);
    convert_x<<<(BATCH * K_IN / 4 + 255) / 256, 256, 0, stream>>>(x, xh);
    gemm1_wmma<<<63, 512, 0, stream>>>(xh, W1, b1, cur1);
    for (int t = 0; t < STEPS; ++t) {
        step_hidden<<<16, 256, 0, stream>>>(cur1, mem1, W2, cur2);
        step_out<<<(BATCH * N_OUT + 127) / 128, 128, 0, stream>>>(cur2, b2, mem2, out, t);
    }
}
